// TransformerBlock_34007551049974
// MI455X (gfx1250) — matrix operand order repaired and checked by reading, not compiled, not run on hardware
//
#include <hip/hip_runtime.h>
#include <cstdint>

#define B_ 2
#define S_ 2048
#define L_ 512
#define WIDTH_ 1024
#define HEADS_ 16
#define CTX_ 768
#define HD_ 64

typedef __attribute__((ext_vector_type(16))) __bf16 bf16x16;
typedef __attribute__((ext_vector_type(8)))  float  f32x8;
typedef unsigned short u16;
typedef unsigned int   u32;
typedef unsigned long long u64;

union Frag16 { u32 u[8]; bf16x16 v; };

__device__ __forceinline__ u16 f2bf(float x) {
  u32 u = __float_as_uint(x);
  u32 r = u + 0x7FFFu + ((u >> 16) & 1u);   // round-to-nearest-even
  return (u16)(r >> 16);
}

__device__ __forceinline__ f32x8 wmma_bf16(bf16x16 a, bf16x16 b, f32x8 c) {
  // 8 args: (neg_a, A, neg_b, B, c_mod, C, reuse_a, reuse_b)
  return __builtin_amdgcn_wmma_f32_16x16x32_bf16(false, a, false, b, (short)0, c, false, false);
}

// Async global -> LDS copy (CDNA5 TDM-lite path, tracked by ASYNCcnt).
// VDST = per-lane LDS byte offset (flat LDS pointer truncates to LDS offset,
// ISA 10.2), VADDR = 64-bit global address, GV mode.
__device__ __forceinline__ void async_g2l_b128(const void* lds, const void* gptr) {
  u32 loff = (u32)(uintptr_t)lds;
  u64 gaddr = (u64)(uintptr_t)gptr;
  asm volatile("global_load_async_to_lds_b128 %0, %1, off"
               :: "v"(loff), "v"(gaddr) : "memory");
}
__device__ __forceinline__ void wait_async0() {
  asm volatile("s_wait_asynccnt 0x0" ::: "memory");
}

// ---------------------------------------------------------------- convert
__global__ __launch_bounds__(256) void conv_f32_bf16(const float* __restrict__ in,
                                                     u16* __restrict__ out, int n) {
  int i = blockIdx.x * 256 + threadIdx.x;
  if (i < n) out[i] = f2bf(in[i]);
}

// ---------------------------------------------------------------- layernorm -> bf16
__global__ __launch_bounds__(256) void ln_bf16_kernel(const float* __restrict__ x,
                                                      const float* __restrict__ g,
                                                      const float* __restrict__ bta,
                                                      u16* __restrict__ out) {
  int row = blockIdx.x, t = threadIdx.x;
  const float* xr = x + (size_t)row * WIDTH_;
  float4 v = *(const float4*)(xr + t * 4);
  float s  = v.x + v.y + v.z + v.w;
  float s2 = v.x * v.x + v.y * v.y + v.z * v.z + v.w * v.w;
  #pragma unroll
  for (int off = 16; off >= 1; off >>= 1) {
    s  += __shfl_xor(s,  off, 32);
    s2 += __shfl_xor(s2, off, 32);
  }
  __shared__ float sh[16];
  if ((t & 31) == 0) { sh[t >> 5] = s; sh[8 + (t >> 5)] = s2; }
  __syncthreads();
  float ts = 0.f, ts2 = 0.f;
  #pragma unroll
  for (int i = 0; i < 8; i++) { ts += sh[i]; ts2 += sh[8 + i]; }
  float mean = ts * (1.0f / WIDTH_);
  float var  = ts2 * (1.0f / WIDTH_) - mean * mean;
  float rinv = rsqrtf(var + 1e-5f);
  int c = t * 4;
  float y0 = (v.x - mean) * rinv * g[c + 0] + bta[c + 0];
  float y1 = (v.y - mean) * rinv * g[c + 1] + bta[c + 1];
  float y2 = (v.z - mean) * rinv * g[c + 2] + bta[c + 2];
  float y3 = (v.w - mean) * rinv * g[c + 3] + bta[c + 3];
  uint2 o;
  o.x = (u32)f2bf(y0) | ((u32)f2bf(y1) << 16);
  o.y = (u32)f2bf(y2) | ((u32)f2bf(y3) << 16);
  *(uint2*)(out + (size_t)row * WIDTH_ + c) = o;
}

// ---------------------------------------------------------------- GEMM: C = A * W^T + bias
// A: [M,K] bf16 row-major.  W: [N,K] bf16 row-major (both K-contiguous -> WMMA friendly).
// Block tile 128x128, k-step 32, double-buffered async global->LDS staging.
// 8 waves: wave (wm,wn) owns a 64x32 tile = 8 WMMA accumulators; per k-step
// 12 ds_load_b128 feed 8 v_wmma (1.5 LDS ops / WMMA).
enum { MODE_RESID = 0, MODE_BF16 = 1, MODE_BF16_GELU = 2, MODE_QK = 3, MODE_VT = 4 };

__global__ __launch_bounds__(256) void gemm_bf16_wmma(
    const u16* __restrict__ A, const u16* __restrict__ W, const float* __restrict__ bias,
    int M, int N, int K, int mode, int Srows,
    const float* __restrict__ resid, float* __restrict__ outF, u16* __restrict__ outB) {
  __shared__ u16 As[2][128 * 32];
  __shared__ u16 Bs[2][128 * 32];
  int tid = threadIdx.x, lane = tid & 31, w = tid >> 5;
  int mBase = blockIdx.y * 128, nBase = blockIdx.x * 128;
  int wm = w >> 2, wn = w & 3;                 // 2 row-groups(64) x 4 col-groups(32)
  int m15 = lane & 15, hi = lane >> 4;

  f32x8 zero = {0, 0, 0, 0, 0, 0, 0, 0};
  f32x8 acc[4][2];
  #pragma unroll
  for (int im = 0; im < 4; im++)
    #pragma unroll
    for (int in = 0; in < 2; in++) acc[im][in] = zero;

  // staging: each thread owns 16 contiguous bf16 (32B) of each 128x32 tile
  int idx = tid * 16;
  int rr = idx >> 5, cc2 = idx & 31;
  const u16* aSrc = A + (size_t)(mBase + rr) * K + cc2;
  const u16* bSrc = W + (size_t)(nBase + rr) * K + cc2;

  // prologue: async-fill buffer 0
  async_g2l_b128(&As[0][idx],     aSrc);
  async_g2l_b128(&As[0][idx + 8], aSrc + 8);
  async_g2l_b128(&Bs[0][idx],     bSrc);
  async_g2l_b128(&Bs[0][idx + 8], bSrc + 8);
  wait_async0();
  __syncthreads();

  int buf = 0;
  for (int kb = 0; kb < K; kb += 32) {
    int nb = buf ^ 1;
    if (kb + 32 < K) {  // async-prefetch next k-slab into the other buffer
      const u16* a2 = aSrc + kb + 32;
      const u16* b2 = bSrc + kb + 32;
      async_g2l_b128(&As[nb][idx],     a2);
      async_g2l_b128(&As[nb][idx + 8], a2 + 8);
      async_g2l_b128(&Bs[nb][idx],     b2);
      async_g2l_b128(&Bs[nb][idx + 8], b2 + 8);
    }
    // A fragments (16x32): lane m holds row m; K = {hi*8..+7, 16+hi*8..+7}
    Frag16 af[4], bf[2];
    #pragma unroll
    for (int im = 0; im < 4; im++) {
      const u16* arow = &As[buf][(wm * 64 + im * 16 + m15) * 32];
      *(uint4*)&af[im].u[0] = *(const uint4*)(arow + hi * 8);
      *(uint4*)&af[im].u[4] = *(const uint4*)(arow + 16 + hi * 8);
    }
    // B fragments (32x16): lane n holds column n; K = {hi*8..+7, 16+hi*8..+7}
    #pragma unroll
    for (int in = 0; in < 2; in++) {
      const u16* brow = &Bs[buf][(wn * 32 + in * 16 + m15) * 32];
      *(uint4*)&bf[in].u[0] = *(const uint4*)(brow + hi * 8);
      *(uint4*)&bf[in].u[4] = *(const uint4*)(brow + 16 + hi * 8);
    }
    #pragma unroll
    for (int im = 0; im < 4; im++)
      #pragma unroll
      for (int in = 0; in < 2; in++)
        acc[im][in] = wmma_bf16(af[im].v, bf[in].v, acc[im][in]);
    wait_async0();
    __syncthreads();
    buf = nb;
  }

  // Epilogue. C layout: VGPR r -> M = (hi ? 8+r : r), N = lane&15.
  #pragma unroll
  for (int im = 0; im < 4; im++) {
    #pragma unroll
    for (int in = 0; in < 2; in++) {
      int col = nBase + wn * 32 + in * 16 + m15;
      float bv = bias[col];
      #pragma unroll
      for (int r = 0; r < 8; r++) {
        float accv = acc[im][in][r] + bv;
        int row = mBase + wm * 64 + im * 16 + (hi ? 8 + r : r);
        if (mode == MODE_RESID) {
          size_t o = (size_t)row * N + col;
          outF[o] = accv + resid[o];
        } else if (mode == MODE_BF16) {
          outB[(size_t)row * N + col] = f2bf(accv);
        } else if (mode == MODE_BF16_GELU) {
          float gl = 0.5f * accv * (1.0f + erff(accv * 0.70710678118f));
          outB[(size_t)row * N + col] = f2bf(gl);
        } else {
          int b = row / Srows, sIdx = row % Srows;
          int h = col >> 6, d = col & 63;
          if (mode == MODE_QK)
            outB[((size_t)(b * HEADS_ + h) * Srows + sIdx) * HD_ + d] = f2bf(accv);
          else  // MODE_VT: V stored transposed [B*H, HD, Skv] for contiguous P@V B-frags
            outB[((size_t)(b * HEADS_ + h) * HD_ + d) * Srows + sIdx] = f2bf(accv);
        }
      }
    }
  }
}

// ---------------------------------------------------------------- flash attention
// Q: [B*H, Sq, 64] bf16;  K: [B*H, Skv, 64] bf16;  VT: [B*H, 64, Skv] bf16
// O: [B, Sq, WIDTH] bf16 (heads merged). One wave per 16-query tile.
__global__ __launch_bounds__(128) void flash_attn_kernel(
    const u16* __restrict__ Q, const u16* __restrict__ Kb, const u16* __restrict__ VT,
    u16* __restrict__ O, int Sq, int Skv, float scale) {
  __shared__ u16 Pls[4][16 * 32];
  int tid = threadIdx.x, lane = tid & 31, w = tid >> 5;
  int gw = blockIdx.x * 4 + w;
  int nqb = Sq >> 4;
  int qb = gw % nqb, bh = gw / nqb;
  int m15 = lane & 15, hi = lane >> 4;

  const u16* Qp = Q  + ((size_t)bh * Sq + qb * 16) * HD_;
  const u16* Kp = Kb + (size_t)bh * Skv * HD_;
  const u16* Vp = VT + (size_t)bh * HD_ * Skv;

  // Q fragments: d 0..31 and 32..63
  Frag16 qf[2];
  const u16* qrow = Qp + m15 * HD_;
  #pragma unroll
  for (int j = 0; j < 2; j++) {
    *(uint4*)&qf[j].u[0] = *(const uint4*)(qrow + j * 32 + hi * 8);
    *(uint4*)&qf[j].u[4] = *(const uint4*)(qrow + j * 32 + 16 + hi * 8);
  }

  f32x8 zero = {0, 0, 0, 0, 0, 0, 0, 0};
  f32x8 o[4];
  float mrow[8], lrow[8];
  #pragma unroll
  for (int g = 0; g < 4; g++) o[g] = zero;
  #pragma unroll
  for (int r = 0; r < 8; r++) { mrow[r] = -3.0e38f; lrow[r] = 0.f; }

  u16* P = &Pls[w][0];

  for (int cb = 0; cb < Skv; cb += 32) {
    // S = Q @ K^T for 32 keys (two 16x16 tiles, two K-steps over d)
    f32x8 s0 = zero, s1 = zero;
    #pragma unroll
    for (int j = 0; j < 2; j++) {
      Frag16 k0, k1;
      const u16* kr0 = Kp + (size_t)(cb + m15) * HD_ + j * 32;
      *(uint4*)&k0.u[0] = *(const uint4*)(kr0 + hi * 8);
      *(uint4*)&k0.u[4] = *(const uint4*)(kr0 + 16 + hi * 8);
      const u16* kr1 = Kp + (size_t)(cb + 16 + m15) * HD_ + j * 32;
      *(uint4*)&k1.u[0] = *(const uint4*)(kr1 + hi * 8);
      *(uint4*)&k1.u[4] = *(const uint4*)(kr1 + 16 + hi * 8);
      s0 = wmma_bf16(qf[j].v, k0.v, s0);
      s1 = wmma_bf16(qf[j].v, k1.v, s1);
    }
    // Online softmax: row m spans lanes within a 16-lane group
    #pragma unroll
    for (int r = 0; r < 8; r++) {
      float v0 = s0[r] * scale, v1 = s1[r] * scale;
      float mx = fmaxf(v0, v1);
      #pragma unroll
      for (int off = 8; off >= 1; off >>= 1) mx = fmaxf(mx, __shfl_xor(mx, off, 32));
      float mn = fmaxf(mrow[r], mx);
      float al = __expf(mrow[r] - mn);
      float p0 = __expf(v0 - mn), p1 = __expf(v1 - mn);
      float sm = p0 + p1;
      #pragma unroll
      for (int off = 8; off >= 1; off >>= 1) sm += __shfl_xor(sm, off, 32);
      lrow[r] = lrow[r] * al + sm;
      mrow[r] = mn;
      #pragma unroll
      for (int g = 0; g < 4; g++) o[g][r] *= al;
      int prow = hi ? (8 + r) : r;
      P[prow * 32 + m15]      = f2bf(p0);
      P[prow * 32 + 16 + m15] = f2bf(p1);
    }
    __builtin_amdgcn_wave_barrier();
    // P (16x32) @ V (32x64): A-frag from LDS, B-frags contiguous from VT
    Frag16 pf;
    const u16* pr = P + m15 * 32;
    *(uint4*)&pf.u[0] = *(const uint4*)(pr + hi * 8);
    *(uint4*)&pf.u[4] = *(const uint4*)(pr + 16 + hi * 8);
    #pragma unroll
    for (int g = 0; g < 4; g++) {
      Frag16 vb;
      const u16* vr = Vp + (size_t)(g * 16 + m15) * Skv + cb;
      *(uint4*)&vb.u[0] = *(const uint4*)(vr + hi * 8);
      *(uint4*)&vb.u[4] = *(const uint4*)(vr + 16 + hi * 8);
      o[g] = wmma_bf16(pf.v, vb.v, o[g]);
    }
    __builtin_amdgcn_wave_barrier();
  }

  // Normalize and write merged-head output
  int b = bh >> 4, h = bh & 15;
  #pragma unroll
  for (int r = 0; r < 8; r++) {
    int row = hi ? 8 + r : r;
    int sIdx = qb * 16 + row;
    float inv = 1.0f / lrow[r];
    #pragma unroll
    for (int g = 0; g < 4; g++) {
      int d = g * 16 + m15;
      O[((size_t)(b * Sq + sIdx)) * WIDTH_ + h * HD_ + d] = f2bf(o[g][r] * inv);
    }
  }
}

// ---------------------------------------------------------------- launcher
extern "C" void kernel_launch(void* const* d_in, const int* in_sizes, int n_in,
                              void* d_out, int out_size, void* d_ws, size_t ws_size,
                              hipStream_t stream) {
  (void)in_sizes; (void)n_in; (void)out_size; (void)ws_size;
  const float* x = (const float*)d_in[0];
  const size_t MS = (size_t)B_ * S_;            // 4096 query rows
  const size_t MC = (size_t)B_ * L_;            // 1024 context rows

  char* p = (char*)d_ws;
  auto take = [&](size_t bytes) { char* r = p; p += (bytes + 255) & ~(size_t)255; return r; };

  float* RES  = (float*)take(MS * WIDTH_ * 4);        // running residual (fp32)
  u16* HLN    = (u16*)take(MS * WIDTH_ * 2);          // LN output bf16
  u16* QB     = (u16*)take(MS * WIDTH_ * 2);          // [B*H, Sq, 64]
  u16* KB     = (u16*)take(MS * WIDTH_ * 2);          // [B*H, Skv, 64]
  u16* VTb    = (u16*)take(MS * WIDTH_ * 2);          // [B*H, 64, Skv]
  u16* ATT    = (u16*)take(MS * WIDTH_ * 2);          // merged heads bf16
  u16* HID    = (u16*)take(MS * 4 * WIDTH_ * 2);      // MLP hidden bf16
  u16* CTXB   = (u16*)take(MC * CTX_ * 2);
  u16* WSAQ = (u16*)take((size_t)WIDTH_ * WIDTH_ * 2);
  u16* WSAK = (u16*)take((size_t)WIDTH_ * WIDTH_ * 2);
  u16* WSAV = (u16*)take((size_t)WIDTH_ * WIDTH_ * 2);
  u16* WSAO = (u16*)take((size_t)WIDTH_ * WIDTH_ * 2);
  u16* WCAQ = (u16*)take((size_t)WIDTH_ * WIDTH_ * 2);
  u16* WCAK = (u16*)take((size_t)WIDTH_ * CTX_ * 2);
  u16* WCAV = (u16*)take((size_t)WIDTH_ * CTX_ * 2);
  u16* WCAO = (u16*)take((size_t)WIDTH_ * WIDTH_ * 2);
  u16* WM1  = (u16*)take((size_t)4 * WIDTH_ * WIDTH_ * 2);
  u16* WM2  = (u16*)take((size_t)4 * WIDTH_ * WIDTH_ * 2);

  auto conv = [&](const void* src, u16* dst, int n) {
    conv_f32_bf16<<<(n + 255) / 256, 256, 0, stream>>>((const float*)src, dst, n);
  };
  auto gemm = [&](const u16* A, const u16* W, const void* bias, int M, int N, int K,
                  int mode, int Srows, const float* resid, float* outF, u16* outB) {
    dim3 g(N / 128, M / 128);
    gemm_bf16_wmma<<<g, 256, 0, stream>>>(A, W, (const float*)bias, M, N, K,
                                          mode, Srows, resid, outF, outB);
  };
  auto flash = [&](const u16* Q, const u16* Kk, const u16* V, u16* O, int Sq, int Skv) {
    int waves = B_ * HEADS_ * (Sq / 16);
    flash_attn_kernel<<<waves / 4, 128, 0, stream>>>(Q, Kk, V, O, Sq, Skv, 0.125f);
  };
  auto ln = [&](const float* xin, const void* g, const void* b, u16* dst) {
    ln_bf16_kernel<<<(int)MS, 256, 0, stream>>>(xin, (const float*)g, (const float*)b, dst);
  };

  // weight / context conversion to bf16
  conv(d_in[8],  WSAQ, WIDTH_ * WIDTH_);
  conv(d_in[10], WSAK, WIDTH_ * WIDTH_);
  conv(d_in[12], WSAV, WIDTH_ * WIDTH_);
  conv(d_in[14], WSAO, WIDTH_ * WIDTH_);
  conv(d_in[16], WCAQ, WIDTH_ * WIDTH_);
  conv(d_in[18], WCAK, WIDTH_ * CTX_);
  conv(d_in[20], WCAV, WIDTH_ * CTX_);
  conv(d_in[22], WCAO, WIDTH_ * WIDTH_);
  conv(d_in[24], WM1,  4 * WIDTH_ * WIDTH_);
  conv(d_in[26], WM2,  4 * WIDTH_ * WIDTH_);
  conv(d_in[1],  CTXB, (int)(MC * CTX_));

  // --- self-attention ---
  ln(x, d_in[2], d_in[3], HLN);
  gemm(HLN, WSAQ, d_in[9],  (int)MS, WIDTH_, WIDTH_, MODE_QK, S_, nullptr, nullptr, QB);
  gemm(HLN, WSAK, d_in[11], (int)MS, WIDTH_, WIDTH_, MODE_QK, S_, nullptr, nullptr, KB);
  gemm(HLN, WSAV, d_in[13], (int)MS, WIDTH_, WIDTH_, MODE_VT, S_, nullptr, nullptr, VTb);
  flash(QB, KB, VTb, ATT, S_, S_);
  gemm(ATT, WSAO, d_in[15], (int)MS, WIDTH_, WIDTH_, MODE_RESID, 1, x, RES, nullptr);

  // --- cross-attention ---
  ln(RES, d_in[4], d_in[5], HLN);
  gemm(HLN,  WCAQ, d_in[17], (int)MS, WIDTH_, WIDTH_, MODE_QK, S_, nullptr, nullptr, QB);
  gemm(CTXB, WCAK, d_in[19], (int)MC, WIDTH_, CTX_,   MODE_QK, L_, nullptr, nullptr, KB);
  gemm(CTXB, WCAV, d_in[21], (int)MC, WIDTH_, CTX_,   MODE_VT, L_, nullptr, nullptr, VTb);
  flash(QB, KB, VTb, ATT, S_, L_);
  gemm(ATT, WCAO, d_in[23], (int)MS, WIDTH_, WIDTH_, MODE_RESID, 1, RES, RES, nullptr);

  // --- MLP ---
  ln(RES, d_in[6], d_in[7], HLN);
  gemm(HLN, WM1, d_in[25], (int)MS, 4 * WIDTH_, WIDTH_, MODE_BF16_GELU, 1, nullptr, nullptr, HID);
  gemm(HID, WM2, d_in[27], (int)MS, WIDTH_, 4 * WIDTH_, MODE_RESID, 1, RES, (float*)d_out, nullptr);
}
